// DPR_81647328297493
// MI455X (gfx1250) — matrix operand order repaired and checked by reading, not compiled, not run on hardware
//
#include <hip/hip_runtime.h>
#include <math.h>

typedef __attribute__((ext_vector_type(16))) __bf16 v16bf;
typedef __attribute__((ext_vector_type(4)))  __bf16 v4bf;
typedef __attribute__((ext_vector_type(8)))  float  v8f;

#define BQ 64
#define PP 128
#define MQd 64
#define MPd 64
#define HD 768
#define DD (MQd * MPd)       // 4096

#define KCH 64               // K-chunk staged per double-buffer step
#define NCHUNK (HD / KCH)    // 12
#define LROW 144             // 128B chunk-row + 16B pad (bank spread, 16B aligned)
#define TSZ (64 * LROW)      // 9216 bytes per staged tile

// ---------- order-preserving float <-> uint keys (ascending) ----------
__device__ __forceinline__ unsigned key_asc(float x) {
    unsigned u = __float_as_uint(x);
    return u ^ (unsigned)(((int)u >> 31) | (int)0x80000000);
}
__device__ __forceinline__ float dec_asc(unsigned k) {
    unsigned u = (k & 0x80000000u) ? (k ^ 0x80000000u) : ~k;
    return __uint_as_float(u);
}

// ---------- CDNA5 async copy: global -> LDS, per-lane 16B, ASYNCcnt ----------
__device__ __forceinline__ void async_load_b128(unsigned lds_addr, const void* gaddr) {
    asm volatile("global_load_async_to_lds_b128 %0, %1, off"
                 :: "v"(lds_addr), "v"(gaddr)
                 : "memory");
}

// ---------- f32 -> bf16 conversion (vectorized x4) ----------
__global__ void cvt_bf16(const float4* __restrict__ in, v4bf* __restrict__ out, int n4) {
    int i = blockIdx.x * blockDim.x + threadIdx.x;
    if (i < n4) {
        float4 f = in[i];
        v4bf o;
        o[0] = (__bf16)f.x; o[1] = (__bf16)f.y;
        o[2] = (__bf16)f.z; o[3] = (__bf16)f.w;
        out[i] = o;
    }
}

// ---------- stage one K-chunk of Q and P into LDS (async, 4 instrs/wave) ----------
__device__ __forceinline__ void stage_chunk(const __bf16* __restrict__ Qg,
                                            const __bf16* __restrict__ Pg,
                                            unsigned ldsQ, unsigned ldsP,
                                            int k0, int tid) {
    #pragma unroll
    for (int j = 0; j < 2; ++j) {                 // Q: 64 rows x 128B = 512 x b128
        int L = j * 256 + tid;                    // 0..511
        int row = L >> 3, segB = (L & 7) * 16;
        async_load_b128(ldsQ + row * LROW + segB,
                        (const char*)(Qg + (size_t)row * HD + k0) + segB);
    }
    #pragma unroll
    for (int j = 0; j < 2; ++j) {                 // P: same shape
        int L = j * 256 + tid;
        int row = L >> 3, segB = (L & 7) * 16;
        async_load_b128(ldsP + row * LROW + segB,
                        (const char*)(Pg + (size_t)row * HD + k0) + segB);
    }
}

// ---------- exact k-th selection (radix, 4x8-bit passes) + masked sum ----------
__device__ float select_sum(const float* __restrict__ S,
                            const unsigned char* __restrict__ qm,
                            const unsigned char* __restrict__ pm,
                            int k, bool desc,
                            int* hist, int* sh_bucket, int* sh_rank, float* sh_sum,
                            int tid) {
    unsigned pref = 0, pbits = 0;
    int r = k;
    for (int shift = 24; shift >= 0; shift -= 8) {
        hist[tid] = 0;
        __syncthreads();
        for (int i = tid; i < DD; i += 256) {
            if (qm[i >> 6] && pm[i & 63]) {
                unsigned key = key_asc(S[i]);
                if (desc) key = ~key;
                if ((key & pbits) == pref)
                    atomicAdd(&hist[(key >> shift) & 255], 1);
            }
        }
        __syncthreads();
        if (tid == 0) {
            int cum = 0, bkt = 255, rem = r;
            for (int j = 0; j < 256; ++j) {
                int c = hist[j];
                if (cum + c >= r) { bkt = j; rem = r - cum; break; }
                cum += c;
            }
            *sh_bucket = bkt;
            *sh_rank   = rem;
        }
        __syncthreads();
        pref |= ((unsigned)*sh_bucket) << shift;
        pbits |= 255u << shift;
        r = *sh_rank;
        __syncthreads();
    }
    if (tid == 0) *sh_sum = 0.f;
    __syncthreads();
    float part = 0.f;
    for (int i = tid; i < DD; i += 256) {
        if (qm[i >> 6] && pm[i & 63]) {
            unsigned key = key_asc(S[i]);
            if (desc) key = ~key;
            if (key < pref) part += S[i];
        }
    }
    atomicAdd(sh_sum, part);
    __syncthreads();
    float thr = dec_asc(desc ? ~pref : pref);
    float result = *sh_sum + (float)r * thr;
    __syncthreads();
    return result;
}

// ---------- fused GEMM (async-LDS staged WMMA bf16) + topk/bottomk scoring ----------
__global__ __launch_bounds__(256) void score_kernel(
    const __bf16* __restrict__ qbf, const __bf16* __restrict__ pbf,
    const unsigned char* __restrict__ qmask, const unsigned char* __restrict__ pmask,
    const float* __restrict__ alpha_raw, const float* __restrict__ beta_raw,
    float* __restrict__ out) {

    // [Q0][P0][Q1][P1] staging buffers; S (16KB) aliases buffer 0 region after GEMM.
    __shared__ __align__(16) char smem[4 * TSZ];
    __shared__ int hist[256];
    __shared__ unsigned char qm[MQd], pm[MPd];
    __shared__ int sh_nq, sh_np, sh_bucket, sh_rank;
    __shared__ float sh_sum;

    float* S = (float*)smem;

    const int p = blockIdx.x, b = blockIdx.y;
    const int tid = threadIdx.x;
    const int wave = tid >> 5, lane = tid & 31;
    const int lh = lane >> 4, lm = lane & 15;

    if (tid == 0) { sh_nq = 0; sh_np = 0; }
    if (tid < MQd)            qm[tid]       = qmask[b * MQd + tid];
    else if (tid < MQd + MPd) pm[tid - MQd] = pmask[p * MPd + (tid - MQd)];

    const __bf16* Qg = qbf + (size_t)b * MQd * HD;
    const __bf16* Pg = pbf + (size_t)p * MPd * HD;
    const unsigned ldsBase = (unsigned)(uintptr_t)(&smem[0]);

    // 8 waves x 2 C-tiles cover the 4x4 grid of 16x16 tiles (A shared per wave).
    const int mt  = wave >> 1;
    const int nt0 = (wave << 1) & 3;
    const int nt1 = nt0 + 1;

    // LDS fragment offsets (chunk-relative).
    // A (16x32 bf16): lane-half holds 16 contiguous K (low: K0-15, high: K16-31).
    const unsigned aoff  = (unsigned)(mt  * 16 + lm) * LROW + lh * 32;
    // B (32x16 bf16): lane-half holds 16 contiguous K (low: K0-15, high: K16-31).
    const unsigned b0off = (unsigned)(nt0 * 16 + lm) * LROW + lh * 32;
    const unsigned b1off = (unsigned)(nt1 * 16 + lm) * LROW + lh * 32;

    v8f c0 = {}; v8f c1 = {};

    stage_chunk(Qg, Pg, ldsBase, ldsBase + TSZ, 0, tid);
    for (int c = 0; c < NCHUNK; ++c) {
        if (c + 1 < NCHUNK) {
            unsigned lq = ldsBase + ((c + 1) & 1) * (2 * TSZ);
            stage_chunk(Qg, Pg, lq, lq + TSZ, (c + 1) * KCH, tid);
            asm volatile("s_wait_asynccnt 4" ::: "memory");   // chunk c landed
        } else {
            asm volatile("s_wait_asynccnt 0" ::: "memory");
        }
        __syncthreads();

        const char* Qb = smem + (c & 1) * (2 * TSZ);
        const char* Pb = Qb + TSZ;
        #pragma unroll
        for (int kk = 0; kk < KCH; kk += 32) {
            union { uint4 u[2]; v16bf v; } a, bA, bB;
            const char* ap  = Qb + aoff  + kk * 2;
            a.u[0]  = *(const uint4*)(ap);
            a.u[1]  = *(const uint4*)(ap + 16);
            const char* bp0 = Pb + b0off + kk * 2;
            bA.u[0] = *(const uint4*)(bp0);
            bA.u[1] = *(const uint4*)(bp0 + 16);
            const char* bp1 = Pb + b1off + kk * 2;
            bB.u[0] = *(const uint4*)(bp1);
            bB.u[1] = *(const uint4*)(bp1 + 16);
            c0 = __builtin_amdgcn_wmma_f32_16x16x32_bf16(false, a.v, false, bA.v,
                                                         (short)0, c0, false, false);
            c1 = __builtin_amdgcn_wmma_f32_16x16x32_bf16(false, a.v, false, bB.v,
                                                         (short)0, c1, false, false);
        }
        __syncthreads();   // protect buffer reuse by the next stage_chunk
    }

    {   // C layout: VGPR r -> M = r (lanes 0-15) or r+8 (lanes 16-31), N = lm
        int row = mt * 16 + 8 * lh;
        #pragma unroll
        for (int r = 0; r < 8; ++r) {
            S[(row + r) * MPd + nt0 * 16 + lm] = c0[r];
            S[(row + r) * MPd + nt1 * 16 + lm] = c1[r];
        }
    }
    __syncthreads();

    // ---- mask counts ----
    if (tid < MQd)            { if (qm[tid])       atomicAdd(&sh_nq, 1); }
    else if (tid < MQd + MPd) { if (pm[tid - MQd]) atomicAdd(&sh_np, 1); }
    __syncthreads();

    const int n  = sh_nq * sh_np;
    const int ns = n > 0 ? n : 1;
    int kk_ = (4 * ns) / 10; kk_ = kk_ < 1 ? 1 : (kk_ > DD ? DD : kk_);
    int ll_ = (2 * ns) / 10; ll_ = ll_ < 1 ? 1 : (ll_ > DD ? DD : ll_);

    // ---- masked total sum ----
    if (tid == 0) sh_sum = 0.f;
    __syncthreads();
    float part = 0.f;
    for (int i = tid; i < DD; i += 256)
        if (qm[i >> 6] && pm[i & 63]) part += S[i];
    atomicAdd(&sh_sum, part);
    __syncthreads();
    float total_sum = sh_sum;
    __syncthreads();

    // ---- exact top-k / bottom-l sums ----
    float top_sum = select_sum(S, qm, pm, kk_, true,  hist, &sh_bucket, &sh_rank, &sh_sum, tid);
    float bot_sum = select_sum(S, qm, pm, ll_, false, hist, &sh_bucket, &sh_rank, &sh_sum, tid);

    if (tid == 0) {
        float logit;
        if (n <= 0) {
            logit = -1e9f;
        } else {
            float total_mean = total_sum / (float)n;
            float top_mean   = top_sum / (float)kk_;
            float bot_mean   = bot_sum / (float)ll_;
            float bneg       = fmaxf(0.f, -bot_mean);
            float alpha      = log1pf(expf(alpha_raw[0]));
            float beta       = log1pf(expf(beta_raw[0]));
            logit = total_mean + alpha * top_mean - beta * bneg;
        }
        out[b * PP + p] = logit;
    }
}

extern "C" void kernel_launch(void* const* d_in, const int* in_sizes, int n_in,
                              void* d_out, int out_size, void* d_ws, size_t ws_size,
                              hipStream_t stream) {
    const float*         q_emb     = (const float*)d_in[0];          // [64,64,768] f32
    const float*         p_emb     = (const float*)d_in[1];          // [128,64,768] f32
    const unsigned char* q_mask    = (const unsigned char*)d_in[2];  // [64,64] bool
    const unsigned char* p_mask    = (const unsigned char*)d_in[3];  // [128,64] bool
    const float*         alpha_raw = (const float*)d_in[4];
    const float*         beta_raw  = (const float*)d_in[5];
    float*               out       = (float*)d_out;                  // [64,128] f32

    const int nq  = BQ * MQd * HD;   // 3,145,728
    const int npp = PP * MPd * HD;   // 6,291,456
    __bf16* qbf = (__bf16*)d_ws;
    __bf16* pbf = qbf + nq;          // total < 19 MB of ws

    const int thr = 256;
    cvt_bf16<<<(nq / 4 + thr - 1) / thr, thr, 0, stream>>>(
        (const float4*)q_emb, (v4bf*)qbf, nq / 4);
    cvt_bf16<<<(npp / 4 + thr - 1) / thr, thr, 0, stream>>>(
        (const float4*)p_emb, (v4bf*)pbf, npp / 4);

    dim3 grid(PP, BQ);   // one workgroup per (b, p) pair
    score_kernel<<<grid, 256, 0, stream>>>(qbf, pbf, q_mask, p_mask,
                                           alpha_raw, beta_raw, out);
}
